// MultiQueryAttention_12979391169022
// MI455X (gfx1250) — matrix operand order repaired and checked by reading, not compiled, not run on hardware
//
#include <hip/hip_runtime.h>

#define S_LEN  4096
#define N_DIMS 1024
#define N_HEADS 16
#define HEAD_D 64

typedef __attribute__((ext_vector_type(16))) __bf16        v16bf;
typedef __attribute__((ext_vector_type(8)))  float         v8f;
typedef __attribute__((ext_vector_type(8)))  unsigned int  v8u;
typedef __attribute__((ext_vector_type(4)))  unsigned int  v4u;
typedef __attribute__((ext_vector_type(8)))  int           v8i;
typedef __attribute__((ext_vector_type(4)))  int           v4i;

#if defined(__HIP_DEVICE_COMPILE__) && __has_builtin(__builtin_amdgcn_tensor_load_to_lds)
#define HAVE_TDM 1
#else
#define HAVE_TDM 0
#endif

// ---------- helpers ----------

__device__ __forceinline__ unsigned short f2bf(float f) {
  unsigned u = __builtin_bit_cast(unsigned, f);
  u += 0x7fffu + ((u >> 16) & 1u);          // round-to-nearest-even
  return (unsigned short)(u >> 16);
}

__device__ __forceinline__ v8f vzero8() {
  v8f z;
  for (int i = 0; i < 8; ++i) z[i] = 0.0f;
  return z;
}

// Load a 16-element bf16 fragment from two 16-byte-aligned LDS chunks
// (CDNA5 wave32 WMMA 16-bit A/B striping: elems 0..7 chunk0, 8..15 chunk1).
__device__ __forceinline__ v16bf frag_ld(const unsigned short* p0,
                                         const unsigned short* p1) {
  const uint4 a = *(const uint4*)p0;
  const uint4 b = *(const uint4*)p1;
  v8u u;
  u[0] = a.x; u[1] = a.y; u[2] = a.z; u[3] = a.w;
  u[4] = b.x; u[5] = b.y; u[6] = b.z; u[7] = b.w;
  return __builtin_bit_cast(v16bf, u);
}

#if HAVE_TDM
__device__ __forceinline__ unsigned lds_off(const void* p) {
  // LDS generic pointers carry the LDS byte offset in the low 32 bits
  return (unsigned)(unsigned long long)(uintptr_t)p;
}

// Issue a 2D TDM tile load: global [tensorDim1 x tensorDim0] (row stride
// stride0, elements of size 2^dataSizeCode bytes) -> LDS tile
// [tileDim1 x tileDim0], optional LDS row padding (D# pad feature).
__device__ __forceinline__ void tdm_load_2d(
    unsigned ldsAddr, const void* gptr, unsigned dataSizeCode,
    unsigned tensorDim0, unsigned tensorDim1, unsigned tileDim0,
    unsigned tileDim1, unsigned long long stride0, unsigned padEnable,
    unsigned padInterval, unsigned padAmount) {
  unsigned long long ga = (unsigned long long)(uintptr_t)gptr;
  v4u g0;
  g0[0] = 1u;                                           // count=1, user mode
  g0[1] = ldsAddr;                                      // lds_addr
  g0[2] = (unsigned)(ga & 0xFFFFFFFFu);                 // global_addr[31:0]
  g0[3] = (unsigned)((ga >> 32) & 0x01FFFFFFu) | (2u << 30);  // ga[56:32], type=2
  v8i g1;
  g1[0] = (int)((dataSizeCode << 16) | (padEnable << 20) |
                (padInterval << 22) | (padAmount << 25));
  g1[1] = (int)((tensorDim0 & 0xFFFFu) << 16);          // tensor_dim0[15:0]
  g1[2] = (int)((tensorDim0 >> 16) | ((tensorDim1 & 0xFFFFu) << 16));
  g1[3] = (int)((tensorDim1 >> 16) | (tileDim0 << 16));
  g1[4] = (int)tileDim1;                                // tile_dim2 = 0 (2D)
  g1[5] = (int)(stride0 & 0xFFFFFFFFull);               // dim0_stride[31:0]
  g1[6] = (int)((stride0 >> 32) & 0xFFFFull);           // dim0_stride[47:32]
  g1[7] = 0;                                            // dim1_stride = 0
  v4i z4 = {0, 0, 0, 0};
#if defined(__clang_major__) && __clang_major__ >= 23
  v8i z8 = {0, 0, 0, 0, 0, 0, 0, 0};
  __builtin_amdgcn_tensor_load_to_lds(g0, g1, z4, z4, z8, 0);
#else
  __builtin_amdgcn_tensor_load_to_lds(g0, g1, z4, z4, 0);
#endif
}
#endif  // HAVE_TDM

// ---------- generic bf16-WMMA GEMM: C = A(f32)[M,K] * B(f32)[K,N] + bias ----------
// Writes f32 (outF) and/or bf16 (outH). M%64==0, N%64==0, K%32==0 assumed.
// 128 threads = 4 waves; block tile 64x64, BK=32; wave tile 32x32 (2x2 WMMA).
// (TDM not usable here: staging converts f32 -> bf16 in flight.)

__global__ __launch_bounds__(128) void gemm_bf16_wmma(
    const float* __restrict__ A, const float* __restrict__ B,
    const float* __restrict__ bias, float* __restrict__ outF,
    unsigned short* __restrict__ outH, int M, int N, int K) {
  // row stride 40 ushorts = 80 B (multiple of 16 -> b128 fragment loads)
  __shared__ unsigned short As[64][40];   // A tile, row-major [m][k]
  __shared__ unsigned short Bt[64][40];   // B tile transposed [n][k]

  const int t  = threadIdx.x;
  const int w  = t >> 5;
  const int l  = t & 31;
  const int g  = l >> 4;       // lane group (K-half / M-half select)
  const int ln = l & 15;
  const int mw = (w >> 1) * 32;
  const int nw = (w & 1) * 32;
  const int m0 = blockIdx.y * 64;
  const int n0 = blockIdx.x * 64;

  v8f acc[2][2];
  for (int i = 0; i < 2; ++i)
    for (int j = 0; j < 2; ++j) acc[i][j] = vzero8();

  for (int k0 = 0; k0 < K; k0 += 32) {
    if (k0 + 32 < K) {  // hint next tiles into cache (global_prefetch_b8)
      __builtin_prefetch(&A[(size_t)(m0 + (t >> 1)) * K + k0 + 32], 0, 1);
      __builtin_prefetch(&B[(size_t)(k0 + 32 + (t & 31)) * N + n0], 0, 1);
    }
    // stage A 64x32 (f32 -> bf16)
    for (int i = 0; i < 16; ++i) {
      int e = t + 128 * i;
      int r = e >> 5, c = e & 31;
      As[r][c] = f2bf(A[(size_t)(m0 + r) * K + k0 + c]);
    }
    // stage B 32x64 transposed (f32 -> bf16)
    for (int i = 0; i < 16; ++i) {
      int e = t + 128 * i;
      int kk = e >> 6, n = e & 63;
      Bt[n][kk] = f2bf(B[(size_t)(k0 + kk) * N + n0 + n]);
    }
    __syncthreads();

    v16bf af[2], bfr[2];
    for (int mi = 0; mi < 2; ++mi) {
      int r = mw + mi * 16 + ln;
      af[mi] = frag_ld(&As[r][g * 8], &As[r][16 + g * 8]);
    }
    for (int ni = 0; ni < 2; ++ni) {
      int n = nw + ni * 16 + ln;
      bfr[ni] = frag_ld(&Bt[n][g * 8], &Bt[n][16 + g * 8]);
    }
    for (int mi = 0; mi < 2; ++mi)
      for (int ni = 0; ni < 2; ++ni)
        acc[mi][ni] = __builtin_amdgcn_wmma_f32_16x16x32_bf16(
            false, af[mi], false, bfr[ni], (short)0, acc[mi][ni], false, false);
    __syncthreads();
  }

  // epilogue: bias + store (C layout: lane col = ln, rows vv + 8*g)
  for (int mi = 0; mi < 2; ++mi)
    for (int ni = 0; ni < 2; ++ni) {
      int col = n0 + nw + ni * 16 + ln;
      float bv = bias ? bias[col] : 0.0f;
      for (int vv = 0; vv < 8; ++vv) {
        int row = m0 + mw + mi * 16 + vv + 8 * g;
        float f = acc[mi][ni][vv] + bv;
        size_t idx = (size_t)row * N + col;
        if (outF) outF[idx] = f;
        if (outH) outH[idx] = f2bf(f);
      }
    }
}

// ---------- flash MQA attention ----------
// Block: 16 query rows x ALL 16 heads (K/V + mask staged once, reused by all
// heads -> mask read exactly once from HBM). 256 threads = 8 waves, 2 heads/wave.
// K tile / mask tile / Q tile staged by the Tensor Data Mover (wave 0 issues,
// s_wait_tensorcnt completes); TDM pad feature reproduces the padded [64][72]
// LDS layout so fragment reads stay ds_load_b128. V is transposed manually
// (TDM can't transpose), overlapping with the in-flight TDM transfers.

__global__ __launch_bounds__(256) void flash_mqa_wmma(
    const unsigned short* __restrict__ Qh,  // [S][DIMS] bf16 (col = h*64+d)
    const unsigned short* __restrict__ Kb,  // [S][64] bf16
    const unsigned short* __restrict__ Vb,  // [S][64] bf16
    const int* __restrict__ mask,           // [S][S] int32
    float* __restrict__ O) {                // [S][DIMS] f32
  __shared__ unsigned short Qs[16][1024];   // Q tile, all heads
  __shared__ unsigned short Ks[64][72];     // K block [key][dim] (B^T for QK^T)
  __shared__ unsigned short Vt[64][72];     // V block transposed [dim][key]
  __shared__ int            MskI[16][64];   // raw mask tile (convert at use)
  __shared__ unsigned short Ps[8][16][72];  // per-wave P bounce (C -> A layout)

  const int t  = threadIdx.x;
  const int w  = t >> 5;
  const int l  = t & 31;
  const int g  = l >> 4;
  const int ln = l & 15;
  const int m0 = blockIdx.x * 16;
  const bool leader = (w == 0);

  // ---- stage Q rows (all heads) ----
#if HAVE_TDM
  if (leader) {
    // 16 x 1024 bf16 tile, contiguous rows (stride 1024), no padding
    tdm_load_2d(lds_off(&Qs[0][0]), Qh + (size_t)m0 * N_DIMS,
                /*dataSize=*/1, /*tDim0=*/N_DIMS, /*tDim1=*/S_LEN,
                /*tileDim0=*/N_DIMS, /*tileDim1=*/16,
                /*stride0=*/N_DIMS, 0, 0, 0);
    __builtin_amdgcn_s_wait_tensorcnt(0);
  }
#else
  for (int i = 0; i < 64; ++i) {
    int e = t + 256 * i;
    int r = e >> 10, c = e & 1023;
    Qs[r][c] = Qh[(size_t)(m0 + r) * N_DIMS + c];
  }
#endif
  __syncthreads();

  // Q fragments, per head, per 32-wide K-dim step
  v16bf qf[2][2];
  for (int hh = 0; hh < 2; ++hh) {
    int h = 2 * w + hh;
    for (int ks = 0; ks < 2; ++ks) {
      int base = h * 64 + ks * 32;
      qf[hh][ks] = frag_ld(&Qs[ln][base + g * 8], &Qs[ln][base + 16 + g * 8]);
    }
  }

  v8f   oacc[2][4];
  float rm[2][8], rl[2][8];
  for (int hh = 0; hh < 2; ++hh) {
    for (int dt = 0; dt < 4; ++dt) oacc[hh][dt] = vzero8();
    for (int vv = 0; vv < 8; ++vv) { rm[hh][vv] = -1e30f; rl[hh][vv] = 0.0f; }
  }

  for (int kb = 0; kb < S_LEN; kb += 64) {
    __syncthreads();
#if HAVE_TDM
    if (leader) {
      // K block: 64 x 64 bf16, rows padded in LDS by 16 B after every 128 B
      // (pad_interval code 4 = 32 dwords, pad_amount code 3 = 4 dwords)
      // -> exactly the [64][72] layout the b128 fragment loads expect.
      tdm_load_2d(lds_off(&Ks[0][0]), Kb + (size_t)kb * HEAD_D,
                  /*dataSize=*/1, /*tDim0=*/HEAD_D, /*tDim1=*/S_LEN,
                  /*tileDim0=*/HEAD_D, /*tileDim1=*/64,
                  /*stride0=*/HEAD_D, /*pad=*/1, /*interval=*/4, /*amount=*/3);
      // mask tile: 16 x 64 int32 slice of the [S][S] mask
      tdm_load_2d(lds_off(&MskI[0][0]), mask + (size_t)m0 * S_LEN + kb,
                  /*dataSize=*/2, /*tDim0=*/S_LEN, /*tDim1=*/S_LEN,
                  /*tileDim0=*/64, /*tileDim1=*/16,
                  /*stride0=*/S_LEN, 0, 0, 0);
    }
#else
    for (int i = 0; i < 16; ++i) {
      int e = t + 256 * i;
      int r = e >> 6, c = e & 63;
      Ks[r][c] = Kb[(size_t)(kb + r) * HEAD_D + c];
    }
    for (int i = 0; i < 4; ++i) {
      int e = t + 256 * i;
      int r = e >> 6, c = e & 63;
      MskI[r][c] = mask[(size_t)(m0 + r) * S_LEN + kb + c];
    }
#endif
    // V block transposed (manual; overlaps TDM transfers)
    for (int i = 0; i < 16; ++i) {
      int e = t + 256 * i;
      int key = e >> 6, d = e & 63;
      Vt[d][key] = Vb[(size_t)(kb + key) * HEAD_D + d];
    }
#if HAVE_TDM
    if (leader) __builtin_amdgcn_s_wait_tensorcnt(0);
#endif
    __syncthreads();

    for (int hh = 0; hh < 2; ++hh) {
      // ---- scores: S = Q_h (16x64) @ K^T (64x64) ----
      v8f sacc[4];
      for (int nt = 0; nt < 4; ++nt) sacc[nt] = vzero8();
      for (int ks = 0; ks < 2; ++ks)
        for (int nt = 0; nt < 4; ++nt) {
          int n = nt * 16 + ln;
          v16bf bk = frag_ld(&Ks[n][ks * 32 + g * 8], &Ks[n][ks * 32 + 16 + g * 8]);
          sacc[nt] = __builtin_amdgcn_wmma_f32_16x16x32_bf16(
              false, qf[hh][ks], false, bk, (short)0, sacc[nt], false, false);
        }

      // ---- mask + online softmax (row r = vv + 8*g lives across 16 lanes) ----
      float nm[8];
      for (int vv = 0; vv < 8; ++vv) {
        float mx = rm[hh][vv];
        for (int nt = 0; nt < 4; ++nt) {
          sacc[nt][vv] += -100000.0f * (float)MskI[vv + 8 * g][nt * 16 + ln];
          mx = fmaxf(mx, sacc[nt][vv]);
        }
        for (int off = 1; off < 16; off <<= 1)
          mx = fmaxf(mx, __shfl_xor(mx, off, 32));
        nm[vv] = mx;
      }
      for (int vv = 0; vv < 8; ++vv) {
        float sc = __expf(rm[hh][vv] - nm[vv]);
        rl[hh][vv] *= sc;
        for (int dt = 0; dt < 4; ++dt) oacc[hh][dt][vv] *= sc;
        float rs = 0.0f;
        for (int nt = 0; nt < 4; ++nt) {
          float p = __expf(sacc[nt][vv] - nm[vv]);
          sacc[nt][vv] = p;
          rs += p;
        }
        for (int off = 1; off < 16; off <<= 1)
          rs += __shfl_xor(rs, off, 32);
        rl[hh][vv] += rs;
        rm[hh][vv] = nm[vv];
      }

      // ---- bounce P through per-wave LDS: C-layout -> A-fragment layout ----
      for (int nt = 0; nt < 4; ++nt)
        for (int vv = 0; vv < 8; ++vv)
          Ps[w][vv + 8 * g][nt * 16 + ln] = f2bf(sacc[nt][vv]);
      asm volatile("s_wait_dscnt 0" ::: "memory");

      // ---- O += P (16x64) @ V (64x64) ----
      for (int ks2 = 0; ks2 < 2; ++ks2) {
        v16bf pa = frag_ld(&Ps[w][ln][ks2 * 32 + g * 8],
                           &Ps[w][ln][ks2 * 32 + 16 + g * 8]);
        for (int dt = 0; dt < 4; ++dt) {
          int n = dt * 16 + ln;
          v16bf bv = frag_ld(&Vt[n][ks2 * 32 + g * 8], &Vt[n][ks2 * 32 + 16 + g * 8]);
          oacc[hh][dt] = __builtin_amdgcn_wmma_f32_16x16x32_bf16(
              false, pa, false, bv, (short)0, oacc[hh][dt], false, false);
        }
      }
    }
  }

  // epilogue: normalize, store f32 attention output [S][h*64+d]
  for (int hh = 0; hh < 2; ++hh) {
    int h = 2 * w + hh;
    for (int dt = 0; dt < 4; ++dt) {
      int col = h * 64 + dt * 16 + ln;
      for (int vv = 0; vv < 8; ++vv) {
        int row = m0 + vv + 8 * g;
        O[(size_t)row * N_DIMS + col] = oacc[hh][dt][vv] / rl[hh][vv];
      }
    }
  }
}

// ---------- launcher ----------

extern "C" void kernel_launch(void* const* d_in, const int* in_sizes, int n_in,
                              void* d_out, int out_size, void* d_ws, size_t ws_size,
                              hipStream_t stream) {
  (void)in_sizes; (void)n_in; (void)out_size; (void)ws_size;
  const float* q   = (const float*)d_in[0];
  const float* k   = (const float*)d_in[1];
  const float* v   = (const float*)d_in[2];
  const int*   msk = (const int*)d_in[3];
  const float* Wq  = (const float*)d_in[4];
  const float* bq  = (const float*)d_in[5];
  const float* Wk  = (const float*)d_in[6];
  const float* bk  = (const float*)d_in[7];
  const float* Wv  = (const float*)d_in[8];
  const float* bv  = (const float*)d_in[9];
  const float* Wo  = (const float*)d_in[10];
  const float* bo  = (const float*)d_in[11];

  float* out = (float*)d_out;                          // [S][DIMS]
  float* khF = out + (size_t)S_LEN * N_DIMS;           // [S][64]  (tuple: kh)
  float* vhF = khF + (size_t)S_LEN * HEAD_D;           // [S][64]  (tuple: vh)

  unsigned short* Qh = (unsigned short*)d_ws;          // [S][DIMS] bf16
  unsigned short* Kb = Qh + (size_t)S_LEN * N_DIMS;    // [S][64]   bf16
  unsigned short* Vb = Kb + (size_t)S_LEN * HEAD_D;    // [S][64]   bf16
  float* O = (float*)(Vb + (size_t)S_LEN * HEAD_D);    // [S][DIMS] f32 attn out

  dim3 blk(128);
  // Q projection -> bf16
  gemm_bf16_wmma<<<dim3(N_DIMS / 64, S_LEN / 64), blk, 0, stream>>>(
      q, Wq, bq, nullptr, Qh, S_LEN, N_DIMS, N_DIMS);
  // K projection -> f32 kh (output tuple) + bf16 copy
  gemm_bf16_wmma<<<dim3(HEAD_D / 64, S_LEN / 64), blk, 0, stream>>>(
      k, Wk, bk, khF, Kb, S_LEN, HEAD_D, N_DIMS);
  // V projection -> f32 vh (output tuple) + bf16 copy
  gemm_bf16_wmma<<<dim3(HEAD_D / 64, S_LEN / 64), blk, 0, stream>>>(
      v, Wv, bv, vhF, Vb, S_LEN, HEAD_D, N_DIMS);
  // flash attention, all heads, mask read once, TDM-staged tiles
  flash_mqa_wmma<<<dim3(S_LEN / 16), dim3(256), 0, stream>>>(Qh, Kb, Vb, msk, O);
  // output projection -> f32 out
  gemm_bf16_wmma<<<dim3(N_DIMS / 64, S_LEN / 64), blk, 0, stream>>>(
      O, Wo, bo, out, nullptr, S_LEN, N_DIMS, N_DIMS);
}
